// MHA_17214228922765
// MI455X (gfx1250) — matrix operand order repaired and checked by reading, not compiled, not run on hardware
//
#include <hip/hip_runtime.h>

// ---------------------------------------------------------------------------
// MHA forward for MI455X (gfx1250): f16 WMMA everywhere, f32 accumulate.
// DIM=1280, HEADS=10, HEAD_DIM=128, BATCH=2, SEQ=2048.
// Projections: 64x64 wave tiles (16 WMMA / K-step, 32 FLOP per L2 byte).
// Attention: causal flash attention, 32 WMMA per 64-key tile per wave.
// ---------------------------------------------------------------------------

typedef _Float16 f16;
typedef __attribute__((ext_vector_type(16))) _Float16 v16h;
typedef __attribute__((ext_vector_type(8)))  _Float16 v8h;
typedef __attribute__((ext_vector_type(8)))  float    v8f;

#define DIMM 1280
#define SEQL 2048
#define NH   10
#define HD   128
#define BT   2
#define MTOT (BT * SEQL)           // 4096 token rows
#define SCALE 0.08838834764831843f // 1/sqrt(128)

#define WMMAF16(a, b, c) \
  __builtin_amdgcn_wmma_f32_16x16x32_f16(false, (a), false, (b), (short)0, (c), false, false)

// ---- fragment loaders (CDNA5 ISA 7.12.2 layouts) ---------------------------

// A-matrix 16x32 f16: lane<16 -> K {kb..kb+7, 16+kb..16+kb+7}, kb=(lane>>4)*8
__device__ __forceinline__ v16h load_frag_a(const f16* __restrict__ p, int ld) {
  int lane = threadIdx.x & 31;
  const f16* q = p + (size_t)(lane & 15) * ld + ((lane >> 4) * 8);
  v8h lo = *(const v8h*)q;
  v8h hi = *(const v8h*)(q + 16);
  return __builtin_shufflevector(lo, hi, 0,1,2,3,4,5,6,7,8,9,10,11,12,13,14,15);
}

// B-matrix 32x16 f16: lane = column N, K {kb..kb+7, 16+kb..16+kb+7}, kb=(lane>>4)*8.
// Source is "row-major over K per column" (weight rows / K rows / V^T rows).
__device__ __forceinline__ v16h load_frag_b(const f16* __restrict__ p, int ld) {
  int lane = threadIdx.x & 31;
  const f16* q = p + (size_t)(lane & 15) * ld + ((lane >> 4) * 8);
  v8h lo = *(const v8h*)q;
  v8h hi = *(const v8h*)(q + 16);
  return __builtin_shufflevector(lo, hi, 0,1,2,3,4,5,6,7,8,9,10,11,12,13,14,15);
}

// ---- f32 -> f16 conversion -------------------------------------------------
__global__ __launch_bounds__(256) void cvt_kernel(const float* __restrict__ in,
                                                  f16* __restrict__ out, int n) {
  int i = blockIdx.x * 256 + threadIdx.x;
  if (i < n) out[i] = (f16)in[i];
}

// ---- 64x64-per-wave GEMM core ----------------------------------------------
// C(64x64) += A(64xK) * B(64xK)^T ; both operands K-major per row.
__device__ __forceinline__ void mma_64x64(const f16* __restrict__ Ab,
                                          const f16* __restrict__ Bb,
                                          int K, v8f (&c)[4][4]) {
  for (int k0 = 0; k0 < K; k0 += 32) {
    if (k0 + 32 < K) {
      // WGP-scope prefetch of the next K-slab (locality 3 -> near caches)
      __builtin_prefetch(Ab + k0 + 64, 0, 3);
      __builtin_prefetch(Bb + k0 + 64, 0, 3);
    }
    v16h a[4], b[4];
    #pragma unroll
    for (int mi = 0; mi < 4; ++mi) a[mi] = load_frag_a(Ab + (size_t)(mi * 16) * K + k0, K);
    #pragma unroll
    for (int ni = 0; ni < 4; ++ni) b[ni] = load_frag_b(Bb + (size_t)(ni * 16) * K + k0, K);
    #pragma unroll
    for (int mi = 0; mi < 4; ++mi)
      #pragma unroll
      for (int ni = 0; ni < 4; ++ni)
        c[mi][ni] = WMMAF16(a[mi], b[ni], c[mi][ni]);
  }
}

// ---- fused QKV projection + RoPE + layout ----------------------------------
// grid: (MTOT/128, DIMM/256, 3)  block: 256 (8 waves as 2(M) x 4(N), 64x64 each)
__global__ __launch_bounds__(256) void qkv_kernel(
    const f16* __restrict__ xh,
    const f16* __restrict__ Wqh, const f16* __restrict__ Wkh, const f16* __restrict__ Wvh,
    const float* __restrict__ bq, const float* __restrict__ bk, const float* __restrict__ bv,
    f16* __restrict__ Qb, f16* __restrict__ Kb, f16* __restrict__ Vtb) {
  int w = threadIdx.x >> 5, lane = threadIdx.x & 31;
  int g = lane >> 4, n = lane & 15;
  int tile_m = blockIdx.x * 128 + (w >> 2) * 64;
  int tile_n = blockIdx.y * 256 + (w & 3) * 64;
  int mode = blockIdx.z;

  const f16* W; const float* bias;
  if (mode == 0)      { W = Wqh; bias = bq; }
  else if (mode == 1) { W = Wkh; bias = bk; }
  else                { W = Wvh; bias = bv; }

  v8f zero = {};
  v8f acc[4][4];
  #pragma unroll
  for (int mi = 0; mi < 4; ++mi)
    #pragma unroll
    for (int ni = 0; ni < 4; ++ni) acc[mi][ni] = zero;

  mma_64x64(xh + (size_t)tile_m * DIMM, W + (size_t)tile_n * DIMM, DIMM, acc);

  #pragma unroll
  for (int mi = 0; mi < 4; ++mi) {
    #pragma unroll
    for (int ni = 0; ni < 4; ++ni) {
      int e   = tile_n + ni * 16 + n;     // output feature column
      int gm0 = tile_m + mi * 16 + g * 8; // first token row this lane holds
      v8f c = acc[mi][ni];
      float be = bias[e];
      int head = e >> 7, dp = e & 127;
      if (mode == 2) {
        // V: store transposed [B,H,128,S] so attention's P@V B-frags are contiguous
        #pragma unroll
        for (int i = 0; i < 8; ++i) {
          int gm = gm0 + i, b = gm >> 11, s = gm & (SEQL - 1);
          Vtb[(((size_t)(b * NH + head)) * HD + dp) * SEQL + s] = (f16)(c[i] + be);
        }
      } else {
        // Q/K: RoPE.  pairs (2j, 2j+1) live in adjacent lanes -> shfl_xor(1)
        f16* out = (mode == 0) ? Qb : Kb;
        int j = dp >> 1;
        float freq = __powf(10000.f, -(float)j * (1.0f / 64.0f));
        bool even = !(dp & 1);
        int dout = head * HD + (even ? j : (64 + j));
        #pragma unroll
        for (int i = 0; i < 8; ++i) {
          int gm = gm0 + i, b = gm >> 11, s = gm & (SEQL - 1);
          float own = c[i] + be;
          float oth = __shfl_xor(own, 1, 32);
          float sn, cs; __sincosf((float)s * freq, &sn, &cs);
          float val = even ? (own * cs - oth * sn)   // x1*cos - x2*sin
                           : (oth * sn + own * cs);  // x1*sin + x2*cos
          out[(((size_t)(b * NH + head)) * SEQL + s) * HD + dout] = (f16)val;
        }
      }
    }
  }
}

// ---- flash attention -------------------------------------------------------
// grid: (SEQ/64, B*H)  block: 128 (4 waves; wave w owns q rows [qtile*64+w*16, +16))
__global__ __launch_bounds__(128) void attn_kernel(
    const f16* __restrict__ Q, const f16* __restrict__ Km,
    const f16* __restrict__ Vt, f16* __restrict__ attnh) {
  __shared__ _Float16 plds[4 * 16 * 64]; // per-wave 16x64 P staging (C->A re-swizzle)
  int w = threadIdx.x >> 5, lane = threadIdx.x & 31;
  int g = lane >> 4, n = lane & 15;
  int qtile = blockIdx.x, bh = blockIdx.y;
  int qr0 = qtile * 64 + w * 16;

  const f16* Qp = Q  + ((size_t)bh * SEQL + qr0) * HD;
  const f16* Kp = Km + (size_t)bh * SEQL * HD;
  const f16* Vp = Vt + (size_t)bh * HD * SEQL;
  f16* pw = plds + w * (16 * 64);

  v16h qf[4];
  #pragma unroll
  for (int kk = 0; kk < 4; ++kk) qf[kk] = load_frag_a(Qp + kk * 32, HD);

  v8f zero = {};
  v8f o[8];
  #pragma unroll
  for (int d = 0; d < 8; ++d) o[d] = zero;
  float mrow[8], lrow[8];
  #pragma unroll
  for (int i = 0; i < 8; ++i) { mrow[i] = -1e30f; lrow[i] = 0.f; }

  for (int kt = 0; kt <= qtile; ++kt) {
    int k0 = kt * 64;
    // ---- scores S = (Q K^T) * scale, 16x64 tile -> 16 WMMAs
    v8f sacc[4];
    #pragma unroll
    for (int nf = 0; nf < 4; ++nf) {
      v8f s = zero;
      #pragma unroll
      for (int kk = 0; kk < 4; ++kk) {
        v16h bf = load_frag_b(Kp + (size_t)(k0 + nf * 16) * HD + kk * 32, HD);
        s = WMMAF16(qf[kk], bf, s);
      }
      sacc[nf] = s;
    }
    // ---- causal mask + scale + running row max
    float mnew[8];
    #pragma unroll
    for (int i = 0; i < 8; ++i) mnew[i] = mrow[i];
    #pragma unroll
    for (int nf = 0; nf < 4; ++nf)
      #pragma unroll
      for (int i = 0; i < 8; ++i) {
        int kidx = k0 + nf * 16 + n;
        int qidx = qr0 + g * 8 + i;
        float sv = sacc[nf][i] * SCALE;
        sv = (kidx > qidx) ? -1e30f : sv;
        sacc[nf][i] = sv;
        mnew[i] = fmaxf(mnew[i], sv);
      }
    #pragma unroll
    for (int x = 1; x < 16; x <<= 1)
      #pragma unroll
      for (int i = 0; i < 8; ++i) mnew[i] = fmaxf(mnew[i], __shfl_xor(mnew[i], x, 32));
    // ---- online softmax update
    float fs[8], rs[8];
    #pragma unroll
    for (int i = 0; i < 8; ++i) { fs[i] = __expf(mrow[i] - mnew[i]); mrow[i] = mnew[i]; rs[i] = 0.f; }
    #pragma unroll
    for (int nf = 0; nf < 4; ++nf)
      #pragma unroll
      for (int i = 0; i < 8; ++i) {
        float p = __expf(sacc[nf][i] - mnew[i]);
        sacc[nf][i] = p;
        rs[i] += p;
      }
    #pragma unroll
    for (int x = 1; x < 16; x <<= 1)
      #pragma unroll
      for (int i = 0; i < 8; ++i) rs[i] += __shfl_xor(rs[i], x, 32);
    #pragma unroll
    for (int i = 0; i < 8; ++i) lrow[i] = lrow[i] * fs[i] + rs[i];
    #pragma unroll
    for (int d = 0; d < 8; ++d)
      #pragma unroll
      for (int i = 0; i < 8; ++i) o[d][i] *= fs[i];
    // ---- P (C-layout) -> LDS -> A-frags (wave-private; LDS is in-order per wave)
    #pragma unroll
    for (int nf = 0; nf < 4; ++nf)
      #pragma unroll
      for (int i = 0; i < 8; ++i)
        pw[(g * 8 + i) * 64 + nf * 16 + n] = (f16)sacc[nf][i];
    v16h pa0 = load_frag_a(pw, 64);
    v16h pa1 = load_frag_a(pw + 32, 64);
    // ---- O += P @ V  (V^T layout: column d contiguous over keys) -> 16 WMMAs
    #pragma unroll
    for (int d = 0; d < 8; ++d) {
      v16h vb0 = load_frag_b(Vp + (size_t)(d * 16) * SEQL + k0, SEQL);
      v16h vb1 = load_frag_b(Vp + (size_t)(d * 16) * SEQL + k0 + 32, SEQL);
      o[d] = WMMAF16(pa0, vb0, o[d]);
      o[d] = WMMAF16(pa1, vb1, o[d]);
    }
  }
  // ---- normalize and store to [B*S, DIM] f16 for the output projection
  int b = bh / NH, h = bh % NH;
  #pragma unroll
  for (int d = 0; d < 8; ++d)
    #pragma unroll
    for (int i = 0; i < 8; ++i) {
      float v = o[d][i] / lrow[i];
      int s = qr0 + g * 8 + i;
      int e = h * HD + d * 16 + n;
      attnh[((size_t)(b * SEQL + s)) * DIMM + e] = (f16)v;
    }
}

// ---- output projection: out = attn @ Wo^T + bo (f32 result) ----------------
// grid: (MTOT/128, DIMM/256)  block: 256
__global__ __launch_bounds__(256) void oproj_kernel(
    const f16* __restrict__ A, const f16* __restrict__ W,
    const float* __restrict__ bias, float* __restrict__ out) {
  int w = threadIdx.x >> 5, lane = threadIdx.x & 31;
  int g = lane >> 4, n = lane & 15;
  int tile_m = blockIdx.x * 128 + (w >> 2) * 64;
  int tile_n = blockIdx.y * 256 + (w & 3) * 64;

  v8f zero = {};
  v8f acc[4][4];
  #pragma unroll
  for (int mi = 0; mi < 4; ++mi)
    #pragma unroll
    for (int ni = 0; ni < 4; ++ni) acc[mi][ni] = zero;

  mma_64x64(A + (size_t)tile_m * DIMM, W + (size_t)tile_n * DIMM, DIMM, acc);

  #pragma unroll
  for (int mi = 0; mi < 4; ++mi)
    #pragma unroll
    for (int ni = 0; ni < 4; ++ni) {
      int e   = tile_n + ni * 16 + n;
      int gm0 = tile_m + mi * 16 + g * 8;
      float be = bias[e];
      #pragma unroll
      for (int i = 0; i < 8; ++i)
        out[(size_t)(gm0 + i) * DIMM + e] = acc[mi][ni][i] + be;
    }
}

// ---------------------------------------------------------------------------
extern "C" void kernel_launch(void* const* d_in, const int* in_sizes, int n_in,
                              void* d_out, int out_size, void* d_ws, size_t ws_size,
                              hipStream_t stream) {
  const float* x  = (const float*)d_in[0];
  // d_in[1] = positions (arange; implied by token index, unused)
  const float* Wq = (const float*)d_in[2]; const float* bq = (const float*)d_in[3];
  const float* Wk = (const float*)d_in[4]; const float* bk = (const float*)d_in[5];
  const float* Wv = (const float*)d_in[6]; const float* bv = (const float*)d_in[7];
  const float* Wo = (const float*)d_in[8]; const float* bo = (const float*)d_in[9];
  float* out = (float*)d_out;

  const size_t NX = (size_t)MTOT * DIMM;        // 5,242,880
  const size_t NW = (size_t)DIMM * DIMM;        // 1,638,400
  const size_t NQ = (size_t)BT * NH * SEQL * HD; // 5,242,880

  f16* ws    = (f16*)d_ws;
  f16* xh    = ws;
  f16* Wqh   = xh  + NX;
  f16* Wkh   = Wqh + NW;
  f16* Wvh   = Wkh + NW;
  f16* Woh   = Wvh + NW;
  f16* Qb    = Woh + NW;
  f16* Kb    = Qb  + NQ;
  f16* Vtb   = Kb  + NQ;
  f16* attnh = Vtb + NQ;

  // 1) f32 -> f16 conversion passes
  cvt_kernel<<<(int)((NX + 255) / 256), 256, 0, stream>>>(x,  xh,  (int)NX);
  cvt_kernel<<<(int)((NW + 255) / 256), 256, 0, stream>>>(Wq, Wqh, (int)NW);
  cvt_kernel<<<(int)((NW + 255) / 256), 256, 0, stream>>>(Wk, Wkh, (int)NW);
  cvt_kernel<<<(int)((NW + 255) / 256), 256, 0, stream>>>(Wv, Wvh, (int)NW);
  cvt_kernel<<<(int)((NW + 255) / 256), 256, 0, stream>>>(Wo, Woh, (int)NW);

  // 2) fused QKV projection + RoPE + V-transpose
  dim3 gq(MTOT / 128, DIMM / 256, 3);
  qkv_kernel<<<gq, 256, 0, stream>>>(xh, Wqh, Wkh, Wvh, bq, bk, bv, Qb, Kb, Vtb);

  // 3) causal flash attention
  dim3 ga(SEQL / 64, BT * NH);
  attn_kernel<<<ga, 128, 0, stream>>>(Qb, Kb, Vtb, attnh);

  // 4) output projection -> f32 d_out
  dim3 go(MTOT / 128, DIMM / 256);
  oproj_kernel<<<go, 256, 0, stream>>>(attnh, Woh, bo, out);
}
